// OTGuidedJointNetwork_32908039422062
// MI455X (gfx1250) — matrix operand order repaired and checked by reading, not compiled, not run on hardware
//
#include <hip/hip_runtime.h>
#include <hip/hip_bf16.h>
#include <math.h>

typedef __attribute__((ext_vector_type(16))) _Float16 v16h;
typedef __attribute__((ext_vector_type(8)))  _Float16 v8h;
typedef __attribute__((ext_vector_type(8)))  float    v8f;

union V16 { v16h v; v8h h[2]; };
union H4  { _Float16 h[4]; unsigned long long u64; };

#define B_  8
#define T_  200
#define U_  50
#define DE  512   // encoder / decoder dim (GEMM K for projections)
#define DF  640   // fusion dim (GEMM N for projections, K for joint)
#define VV  1024  // vocab (GEMM N for joint)

// ---------------------------------------------------------------------------
// Weight convert+transpose: src[K][N] f32 (row-major) -> dst[N][K] f16.
// K-major f16 layout makes each WMMA B-fragment two contiguous b128 loads.
// ---------------------------------------------------------------------------
__global__ void wcvt_kernel(const float* __restrict__ src, _Float16* __restrict__ dst,
                            int K, int N) {
  int i = blockIdx.x * blockDim.x + threadIdx.x;
  if (i >= K * N) return;
  int k = i / N, n = i % N;
  dst[(size_t)n * K + k] = (_Float16)src[i];
}

// ---------------------------------------------------------------------------
// Projection GEMM: out[r][0..DF) = sum_ai A_ai[r][0..DE) @ W_ai + b1 (+ b2)
// Block: 256 threads = 8 waves (2x4), tile 64 rows x 128 cols, col-loop over DF.
// A staged in LDS as f16 (converted from f32 global).
// ---------------------------------------------------------------------------
__global__ __launch_bounds__(256) void proj_kernel(
    const float* __restrict__ A1, const _Float16* __restrict__ Wt1, const float* __restrict__ b1,
    const float* __restrict__ A2, const _Float16* __restrict__ Wt2, const float* __restrict__ b2,
    int M, float* __restrict__ out)
{
  extern __shared__ _Float16 sm[];           // [nA][64][DE] f16
  const int tid = threadIdx.x;
  const int r0  = blockIdx.x * 64;
  const int nA  = (A2 != nullptr) ? 2 : 1;

  for (int ai = 0; ai < nA; ++ai) {
    const float* A = ai ? A2 : A1;
    _Float16* dst = sm + ai * (64 * DE);
    for (int q = tid; q < 64 * DE / 4; q += 256) {
      int row = q / (DE / 4);
      int k   = (q % (DE / 4)) * 4;
      int r   = r0 + row;
      float4 v = make_float4(0.f, 0.f, 0.f, 0.f);
      if (r < M) v = *(const float4*)(A + (size_t)r * DE + k);
      H4 p;
      p.h[0] = (_Float16)v.x; p.h[1] = (_Float16)v.y;
      p.h[2] = (_Float16)v.z; p.h[3] = (_Float16)v.w;
      *(unsigned long long*)(dst + row * DE + k) = p.u64;
    }
  }
  __syncthreads();

  const int lane = tid & 31;
  const int w    = tid >> 5;
  const int wr   = (w >> 2) * 32;   // wave row offset within 64
  const int wc   = (w & 3) * 32;    // wave col offset within 128
  const int l16  = lane & 15;
  const int hi   = lane >> 4;       // 0: lanes 0-15, 1: lanes 16-31

  for (int n0 = 0; n0 < DF; n0 += 128) {
    v8f acc[2][2] = {};
    for (int ai = 0; ai < nA; ++ai) {
      const _Float16* As = sm + ai * (64 * DE);
      const _Float16* Wt = ai ? Wt2 : Wt1;
      for (int k0 = 0; k0 < DE; k0 += 32) {
        V16 a[2], bb[2];
        #pragma unroll
        for (int mi = 0; mi < 2; ++mi) {      // A 16x32: lane<16 K{0..7,16..23}, lane>=16 +8
          const _Float16* p = As + (wr + mi * 16 + l16) * DE + k0 + hi * 8;
          a[mi].h[0] = *(const v8h*)(p);
          a[mi].h[1] = *(const v8h*)(p + 16);
        }
        #pragma unroll
        for (int ni = 0; ni < 2; ++ni) {      // B 32x16: lane=N, lane<16 K{0..7,16..23}, lane>=16 +8
          const _Float16* p = Wt + (size_t)(n0 + wc + ni * 16 + l16) * DE + k0 + hi * 8;
          bb[ni].h[0] = *(const v8h*)(p);
          bb[ni].h[1] = *(const v8h*)(p + 16);
        }
        #pragma unroll
        for (int mi = 0; mi < 2; ++mi)
          #pragma unroll
          for (int ni = 0; ni < 2; ++ni)
            acc[mi][ni] = __builtin_amdgcn_wmma_f32_16x16x32_f16(
                false, a[mi].v, false, bb[ni].v, (short)0, acc[mi][ni], false, false);
      }
    }
    #pragma unroll
    for (int ni = 0; ni < 2; ++ni) {
      int col = n0 + wc + ni * 16 + l16;
      float bias = b1[col] + ((nA == 2) ? b2[col] : 0.f);
      #pragma unroll
      for (int mi = 0; mi < 2; ++mi) {
        #pragma unroll
        for (int j = 0; j < 8; ++j) {         // C/D: lane<16 -> M=j, lane>=16 -> M=j+8
          int row = r0 + wr + mi * 16 + hi * 8 + j;
          if (row < M) out[(size_t)row * DF + col] = acc[mi][ni][j] + bias;
        }
      }
    }
  }
}

// ---------------------------------------------------------------------------
// Joint GEMM: out[r][v] = tanh(fused[b,t]+pdec[b,u]) @ WtOut + b_out
// r = (b*T + t)*U + u, 80000 rows = 1250 blocks x 64 rows. tanh'd f16 A tile
// staged once in 80KB LDS, reused across all 8 vocab col-tiles.
// ---------------------------------------------------------------------------
__global__ __launch_bounds__(256) void OTGuidedJointNetwork_joint_kernel(
    const float* __restrict__ fused, const float* __restrict__ pdec,
    const _Float16* __restrict__ Wt, const float* __restrict__ bout,
    float* __restrict__ out)
{
  extern __shared__ _Float16 As[];           // [64][DF] f16 = 80KB
  const int tid = threadIdx.x;
  const long long r0 = (long long)blockIdx.x * 64;

  for (int q = tid; q < 64 * DF / 4; q += 256) {
    int row = q / (DF / 4);
    int k   = (q % (DF / 4)) * 4;
    long long r = r0 + row;
    int b   = (int)(r / (T_ * U_));
    int rem = (int)(r % (T_ * U_));
    int t   = rem / U_;
    int u   = rem % U_;
    const float4 f = *(const float4*)(fused + ((size_t)b * T_ + t) * DF + k);
    const float4 d = *(const float4*)(pdec  + ((size_t)b * U_ + u) * DF + k);
    H4 p;
    p.h[0] = (_Float16)tanhf(f.x + d.x);
    p.h[1] = (_Float16)tanhf(f.y + d.y);
    p.h[2] = (_Float16)tanhf(f.z + d.z);
    p.h[3] = (_Float16)tanhf(f.w + d.w);
    *(unsigned long long*)(As + row * DF + k) = p.u64;
  }
  __syncthreads();

  const int lane = tid & 31;
  const int w    = tid >> 5;
  const int wr   = (w >> 2) * 32;
  const int wc   = (w & 3) * 32;
  const int l16  = lane & 15;
  const int hi   = lane >> 4;

  for (int n0 = 0; n0 < VV; n0 += 128) {
    v8f acc[2][2] = {};
    for (int k0 = 0; k0 < DF; k0 += 32) {
      V16 a[2], bb[2];
      #pragma unroll
      for (int mi = 0; mi < 2; ++mi) {
        const _Float16* p = As + (wr + mi * 16 + l16) * DF + k0 + hi * 8;
        a[mi].h[0] = *(const v8h*)(p);
        a[mi].h[1] = *(const v8h*)(p + 16);
      }
      #pragma unroll
      for (int ni = 0; ni < 2; ++ni) {
        const _Float16* p = Wt + (size_t)(n0 + wc + ni * 16 + l16) * DF + k0 + hi * 8;
        bb[ni].h[0] = *(const v8h*)(p);
        bb[ni].h[1] = *(const v8h*)(p + 16);
      }
      #pragma unroll
      for (int mi = 0; mi < 2; ++mi)
        #pragma unroll
        for (int ni = 0; ni < 2; ++ni)
          acc[mi][ni] = __builtin_amdgcn_wmma_f32_16x16x32_f16(
              false, a[mi].v, false, bb[ni].v, (short)0, acc[mi][ni], false, false);
    }
    #pragma unroll
    for (int ni = 0; ni < 2; ++ni) {
      int col = n0 + wc + ni * 16 + l16;
      float bias = bout[col];
      #pragma unroll
      for (int mi = 0; mi < 2; ++mi) {
        #pragma unroll
        for (int j = 0; j < 8; ++j) {
          long long row = r0 + wr + mi * 16 + hi * 8 + j;
          __builtin_nontemporal_store(acc[mi][ni][j] + bias,
                                      out + (size_t)row * VV + col);
        }
      }
    }
  }
}

extern "C" void kernel_launch(void* const* d_in, const int* in_sizes, int n_in,
                              void* d_out, int out_size, void* d_ws, size_t ws_size,
                              hipStream_t stream) {
  const float* enc   = (const float*)d_in[0];
  const float* dec   = (const float*)d_in[1];
  const float* ot    = (const float*)d_in[2];
  const float* W_enc = (const float*)d_in[3];
  const float* b_enc = (const float*)d_in[4];
  const float* W_ot  = (const float*)d_in[5];
  const float* b_ot  = (const float*)d_in[6];
  const float* W_dec = (const float*)d_in[7];
  const float* b_dec = (const float*)d_in[8];
  const float* W_out = (const float*)d_in[9];
  const float* b_out = (const float*)d_in[10];
  float* out = (float*)d_out;

  // Workspace layout (~8.4 MB total):
  _Float16* WtEnc = (_Float16*)d_ws;                 // [DF][DE] f16
  _Float16* WtOt  = WtEnc + (size_t)DE * DF;
  _Float16* WtDec = WtOt  + (size_t)DE * DF;
  _Float16* WtOut = WtDec + (size_t)DE * DF;         // [VV][DF] f16
  float* fused = (float*)(WtOut + (size_t)DF * VV);  // [B*T][DF] f32
  float* pdec  = fused + (size_t)(B_ * T_) * DF;     // [B*U][DF] f32

  wcvt_kernel<<<(DE * DF + 255) / 256, 256, 0, stream>>>(W_enc, WtEnc, DE, DF);
  wcvt_kernel<<<(DE * DF + 255) / 256, 256, 0, stream>>>(W_ot,  WtOt,  DE, DF);
  wcvt_kernel<<<(DE * DF + 255) / 256, 256, 0, stream>>>(W_dec, WtDec, DE, DF);
  wcvt_kernel<<<(DF * VV + 255) / 256, 256, 0, stream>>>(W_out, WtOut, DF, VV);

  proj_kernel<<<(B_ * T_ + 63) / 64, 256, 2 * 64 * DE * sizeof(_Float16), stream>>>(
      enc, WtEnc, b_enc, ot, WtOt, b_ot, B_ * T_, fused);
  proj_kernel<<<(B_ * U_ + 63) / 64, 256, 1 * 64 * DE * sizeof(_Float16), stream>>>(
      dec, WtDec, b_dec, nullptr, nullptr, nullptr, B_ * U_, pdec);

  OTGuidedJointNetwork_joint_kernel<<<(B_ * T_ * U_) / 64, 256,
      64 * DF * sizeof(_Float16), stream>>>(fused, pdec, WtOut, b_out, out);
}
